// MentionScore_60224031424678
// MI455X (gfx1250) — matrix operand order repaired and checked by reading, not compiled, not run on hardware
//
#include <hip/hip_runtime.h>
#include <hip/hip_bf16.h>

typedef __attribute__((ext_vector_type(16))) __bf16 v16bf;
typedef __attribute__((ext_vector_type(4)))  __bf16 v4bf;
typedef __attribute__((ext_vector_type(8)))  float  v8f;
typedef __attribute__((ext_vector_type(4)))  float  v4f;
typedef __attribute__((ext_vector_type(4)))  unsigned v4u;
typedef __attribute__((ext_vector_type(8)))  unsigned v8u;

#define HID 150
#define NT 10            // 160 / 16 output column tiles
#define GEMM_BLOCK 320   // 10 waves of 32

__device__ __forceinline__ __bf16 to_bf16(float v) { return (__bf16)v; }

// ---------------------------------------------------------------------------
// Pack f32 weights [Kreal x 150] (row-major, fan_in x fan_out) into bf16
// WMMA B-fragments, padded to [Ktiles*32 x 160].
// B (KxN, 16-bit) layout matching A: lanes 0-15 hold K=0..7,16..23 (j=0..15),
// lanes 16-31 hold K=8..15,24..31, column n = lane & 15.  frag[kt][nt][lane*16+j]
// ---------------------------------------------------------------------------
__global__ void pack_w_kernel(const float* __restrict__ w, int kreal, int ktiles,
                              __bf16* __restrict__ frag) {
  int gid = blockIdx.x * blockDim.x + threadIdx.x;
  int total = ktiles * NT * 512;
  if (gid >= total) return;
  int elem = gid & 511;
  int nt   = (gid >> 9) % NT;
  int kt   = gid / (512 * NT);
  int l = elem >> 4, j = elem & 15;
  int col = nt * 16 + (l & 15);
  int k   = kt * 32 + ((l < 16) ? 0 : 8) + ((j < 8) ? j : j + 8);
  float v = (col < HID && k < kreal) ? w[(size_t)k * HID + col] : 0.0f;
  frag[gid] = to_bf16(v);
}

// ---------------------------------------------------------------------------
// Layer-1 GEMM: C[M x 160] = relu( A[M x lda](f32) * W + b ), output written
// as bf16 A-fragments for the next layer ([mtiles][5][512], K2 = 160).
// grid.x = M/16, block = 320 (10 waves; wave w owns column tile w).
//
// A-fragment (16x32, 16-bit) layout per ISA 7.12.2:
//   lane<16 : M=lane,    j=0..7 -> K=0..7,  j=8..15 -> K=16..23
//   lane>=16: M=lane-16, j=0..7 -> K=8..15, j=8..15 -> K=24..31
// A float4 at (m, 4c) maps to ONE lane, 4 consecutive j -> one ds_store_b64.
// 128 threads stage the 16x32 tile; double-buffered LDS + register prefetch
// (clamped k index so the staging region is a single EXEC-masked block).
// ---------------------------------------------------------------------------
__global__ void gemm_l1_kernel(const float* __restrict__ A, int lda, int ktiles,
                               const __bf16* __restrict__ wfrag,
                               const float* __restrict__ bias,
                               __bf16* __restrict__ outFrag) {
  __shared__ alignas(32) __bf16 aStage[2][512];
  const int wave  = threadIdx.x >> 5;
  const int lane  = threadIdx.x & 31;
  const int mtile = blockIdx.x;

  // staging geometry (threads 0..127)
  const bool staging = threadIdx.x < 128;
  const int sm = (threadIdx.x & 127) >> 3;   // row 0..15
  const int k0 = (threadIdx.x & 7) * 4;      // col group 0,4,..,28
  const int lanep = sm + (((k0 & 15) >= 8) ? 16 : 0);
  const int jp    = (k0 & 7) + ((k0 >= 16) ? 8 : 0);
  const float* aRow = A + (size_t)(mtile * 16 + sm) * lda + k0;

  v4f cur = {};
  if (staging) cur = *(const v4f*)aRow;

  v8f acc = {};
  for (int kt = 0; kt < ktiles; ++kt) {
    const int ktn = (kt + 1 < ktiles) ? kt + 1 : kt;  // clamped prefetch index
    v4f nxt = cur;
    if (staging) {
      v4bf h;
      h[0] = to_bf16(cur[0]); h[1] = to_bf16(cur[1]);
      h[2] = to_bf16(cur[2]); h[3] = to_bf16(cur[3]);
      *(v4bf*)&aStage[kt & 1][lanep * 16 + jp] = h;
      nxt = *(const v4f*)(aRow + (size_t)ktn * 32);   // overlaps with WMMA below
    }
    // WGP-scope prefetch of next B fragment (global_prefetch_b8)
    __builtin_prefetch(&wfrag[((size_t)ktn * NT + wave) * 512 + (lane << 4)], 0, 3);
    __syncthreads();
    v16bf a = *(const v16bf*)(&aStage[kt & 1][lane << 4]);
    v16bf b = *(const v16bf*)(&wfrag[((size_t)kt * NT + wave) * 512 + (lane << 4)]);
    acc = __builtin_amdgcn_wmma_f32_16x16x32_bf16(false, a, false, b, (short)0, acc,
                                                  false, false);
    cur = nxt;
  }
  // Epilogue: bias + relu, scatter into next layer's A-fragment layout.
  // D layout: VGPR r, lanes 0-15 -> (M=r, N=lane); lanes 16-31 -> (M=r+8, N=lane-16)
  int col = wave * 16 + (lane & 15);
  float bv = (col < HID) ? bias[col] : 0.0f;
  int k2t = col >> 5;
  int kk  = col & 31;
  int jo  = (kk & 7) + ((kk >= 16) ? 8 : 0);
  int lhi = ((kk & 15) >= 8) ? 16 : 0;
  __bf16* dst = outFrag + ((size_t)mtile * 5 + k2t) * 512;
  for (int r = 0; r < 8; ++r) {
    int m = r + ((lane >= 16) ? 8 : 0);
    float v = fmaxf(acc[r] + bv, 0.0f);
    dst[(m + lhi) * 16 + jo] = to_bf16(v);
  }
}

// ---------------------------------------------------------------------------
// Layer-2 GEMM: H2[M x 160] = relu( H1frag * W2 + b ), H1 already in
// A-fragment layout, K = 160 (5 k-tiles). Output f32 row-major for layer 3.
//  - W2 fragments (50 KB contiguous) are DMA'd to LDS by wave 0 with ONE
//    Tensor Data Mover descriptor (tensor_load_to_lds, TENSORcnt).
//  - This block's A-fragments (5 KB) are copied via the async-to-LDS engine
//    (ASYNCcnt), one global_load_async_to_lds_b128 per thread, so all 10
//    waves read A from LDS instead of 10x from global.
// ---------------------------------------------------------------------------
__global__ void gemm_l2_kernel(const __bf16* __restrict__ aFrag,
                               const __bf16* __restrict__ wfrag,
                               const float* __restrict__ bias,
                               float* __restrict__ H2) {
  __shared__ alignas(16) __bf16 wLds[5 * NT * 512];  // 25600 elems = 50 KB
  __shared__ alignas(16) __bf16 aLds[5 * 512];       // 2560 elems  = 5 KB
  const int wave  = threadIdx.x >> 5;
  const int lane  = threadIdx.x & 31;
  const int mtile = blockIdx.x;

  if (threadIdx.x < 32) {
    // Build D# (groups 0 + 1) for a 1-row copy: 6400 x 8B = 51200 bytes.
    unsigned ldsW = (unsigned)(uintptr_t)(void*)wLds;
    unsigned long long ga = (unsigned long long)(uintptr_t)wfrag;
    v4u g0;
    g0[0] = 1u;                                   // count=1, user mode
    g0[1] = ldsW;                                 // lds_addr
    g0[2] = (unsigned)ga;                         // global_addr[31:0]
    g0[3] = (unsigned)((ga >> 32) & 0x01FFFFFFu)  // global_addr[56:32]
          | (2u << 30);                           // type=2 (image)
    v8u g1;
    g1[0] = (3u << 16);            // workgroup_mask=0, data_size=3 (8B)
    g1[1] = (6400u & 0xFFFFu) << 16;              // tensor_dim0[15:0]
    g1[2] = (6400u >> 16) | (1u << 16);           // tensor_dim0[31:16] | tensor_dim1=1
    g1[3] = (6400u << 16);                        // tensor_dim1[31:16]=0 | tile_dim0=6400
    g1[4] = 1u;                                   // tile_dim1=1, tile_dim2=0
    g1[5] = 6400u;                                // tensor_dim0_stride
    g1[6] = 0u;
    g1[7] = 0u;
    asm volatile("tensor_load_to_lds %0, %1" :: "s"(g0), "s"(g1) : "memory");
    __builtin_amdgcn_s_wait_tensorcnt(0);
  }

  // async copy this m-tile's A-fragments: 320 x b128 = 5120 bytes
  {
    unsigned ldsA = (unsigned)(uintptr_t)(void*)aLds;
    const char* g = (const char*)(aFrag + (size_t)mtile * 5 * 512) +
                    (size_t)threadIdx.x * 16;
    unsigned l = ldsA + (unsigned)threadIdx.x * 16u;
    asm volatile("global_load_async_to_lds_b128 %0, %1, off"
                 :: "v"(l), "v"(g) : "memory");
  }
  asm volatile("s_wait_asynccnt 0x0" ::: "memory");
  __syncthreads();

  v8f acc = {};
  for (int kt = 0; kt < 5; ++kt) {
    v16bf a = *(const v16bf*)(&aLds[(size_t)kt * 512 + (lane << 4)]);
    v16bf b = *(const v16bf*)(&wLds[((size_t)kt * NT + wave) * 512 + (lane << 4)]);
    acc = __builtin_amdgcn_wmma_f32_16x16x32_bf16(false, a, false, b, (short)0, acc,
                                                  false, false);
  }
  int col = wave * 16 + (lane & 15);
  float bv = (col < HID) ? bias[col] : 0.0f;
  for (int r = 0; r < 8; ++r) {
    int m = mtile * 16 + r + ((lane >= 16) ? 8 : 0);
    H2[(size_t)m * 160 + col] = fmaxf(acc[r] + bv, 0.0f);
  }
}

// ---------------------------------------------------------------------------
// Layer-3: out[m] = H2[m, 0:150] . w3 + b3
// ---------------------------------------------------------------------------
__global__ void dot_l3_kernel(const float* __restrict__ H2, const float* __restrict__ w3,
                              const float* __restrict__ b3, float* __restrict__ out,
                              int M) {
  int m = blockIdx.x * blockDim.x + threadIdx.x;
  if (m >= M) return;
  float s = b3[0];
  const float* row = H2 + (size_t)m * 160;
  for (int k = 0; k < HID; ++k) s += row[k] * w3[k];
  out[m] = s;
}

// ---------------------------------------------------------------------------
// Prefix sum of weighted = embeds * score over T (chunked 3-phase scan).
// Grid decode: bid -> b (8) x tchunk (8) x echunk (4); 256 threads over e.
// ---------------------------------------------------------------------------
__global__ void prefix_partial_kernel(const float* __restrict__ embeds,
                                      const float* __restrict__ score,
                                      float* __restrict__ csum) {
  int bid = blockIdx.x;
  int b = bid >> 5, rem = bid & 31;
  int tc = rem >> 2, ec = rem & 3;
  int e = ec * 256 + threadIdx.x;
  float s = 0.f;
  int t0 = tc * 256;
  for (int t = t0; t < t0 + 256; ++t)
    s += embeds[((size_t)b * 2048 + t) * 1024 + e] * score[b * 2048 + t];
  csum[((size_t)b * 8 + tc) * 1024 + e] = s;
}

__global__ void prefix_scan_kernel(float* __restrict__ csum) {
  int gid = blockIdx.x * blockDim.x + threadIdx.x;  // 8192 = 8 * 1024
  if (gid >= 8192) return;
  int b = gid >> 10, e = gid & 1023;
  float run = 0.f;
  for (int c = 0; c < 8; ++c) {
    size_t i = ((size_t)b * 8 + c) * 1024 + e;
    float v = csum[i];
    csum[i] = run;       // exclusive chunk offset
    run += v;
  }
}

__global__ void prefix_write_kernel(const float* __restrict__ embeds,
                                    const float* __restrict__ score,
                                    const float* __restrict__ csum,
                                    float* __restrict__ prefix) {
  int bid = blockIdx.x;
  int b = bid >> 5, rem = bid & 31;
  int tc = rem >> 2, ec = rem & 3;
  int e = ec * 256 + threadIdx.x;
  float run = csum[((size_t)b * 8 + tc) * 1024 + e];
  if (tc == 0) prefix[(size_t)b * 2049 * 1024 + e] = 0.f;
  int t0 = tc * 256;
  for (int t = t0; t < t0 + 256; ++t) {
    run += embeds[((size_t)b * 2048 + t) * 1024 + e] * score[b * 2048 + t];
    prefix[((size_t)b * 2049 + t + 1) * 1024 + e] = run;
  }
}

// ---------------------------------------------------------------------------
// Span gather: out[b,s] = [embeds[start] | embeds[end] | prefix[end+1]-prefix[start]]
// ---------------------------------------------------------------------------
__global__ void gather_kernel(const float* __restrict__ embeds,
                              const float* __restrict__ prefix,
                              const int* __restrict__ starts,
                              const int* __restrict__ lens,
                              float* __restrict__ out) {
  int span = blockIdx.x;         // 16000 = 8 * 2000
  int b = span / 2000;
  int st = starts[span];
  int en = st + lens[span];
  size_t ob = (size_t)span * 3072;
  for (int e = threadIdx.x; e < 1024; e += 256) {
    float se = embeds[((size_t)b * 2048 + st) * 1024 + e];
    float ee = embeds[((size_t)b * 2048 + en) * 1024 + e];
    float ss = prefix[((size_t)b * 2049 + en + 1) * 1024 + e]
             - prefix[((size_t)b * 2049 + st) * 1024 + e];
    out[ob + e]        = se;
    out[ob + 1024 + e] = ee;
    out[ob + 2048 + e] = ss;
  }
}

// ---------------------------------------------------------------------------
extern "C" void kernel_launch(void* const* d_in, const int* in_sizes, int n_in,
                              void* d_out, int out_size, void* d_ws, size_t ws_size,
                              hipStream_t stream) {
  (void)in_sizes; (void)n_in; (void)out_size; (void)ws_size;
  const float* embeds = (const float*)d_in[0];
  const float* a_w1 = (const float*)d_in[1];
  const float* a_b1 = (const float*)d_in[2];
  const float* a_w2 = (const float*)d_in[3];
  const float* a_b2 = (const float*)d_in[4];
  const float* a_w3 = (const float*)d_in[5];
  const float* a_b3 = (const float*)d_in[6];
  const float* s_w1 = (const float*)d_in[7];
  const float* s_b1 = (const float*)d_in[8];
  const float* s_w2 = (const float*)d_in[9];
  const float* s_b2 = (const float*)d_in[10];
  const float* s_w3 = (const float*)d_in[11];
  const float* s_b3 = (const float*)d_in[12];
  const int* starts = (const int*)d_in[13];
  const int* lens   = (const int*)d_in[14];
  float* out = (float*)d_out;

  char* ws = (char*)d_ws;
  size_t off = 0;
  auto alloc = [&](size_t bytes) -> char* {
    char* p = ws + off;
    off = (off + bytes + 255) & ~(size_t)255;
    return p;
  };
  float*  scoreA = (float*)alloc((size_t)16384 * 4);
  __bf16* w1aF = (__bf16*)alloc((size_t)32 * NT * 512 * 2);
  __bf16* w2aF = (__bf16*)alloc((size_t)5  * NT * 512 * 2);
  __bf16* w1sF = (__bf16*)alloc((size_t)96 * NT * 512 * 2);
  __bf16* w2sF = (__bf16*)alloc((size_t)5  * NT * 512 * 2);
  __bf16* h1F  = (__bf16*)alloc((size_t)1024 * 5 * 512 * 2);   // reused attn -> mention
  float*  h2   = (float*) alloc((size_t)16384 * 160 * 4);      // reused attn -> mention
  float*  pre  = (float*) alloc((size_t)8 * 2049 * 1024 * 4);
  float*  csum = (float*) alloc((size_t)8 * 8 * 1024 * 4);

  // ---- pack MLP weights into bf16 WMMA B-fragments (L2-resident) ----
  int tw;
  tw = 32 * NT * 512; pack_w_kernel<<<(tw + 255) / 256, 256, 0, stream>>>(a_w1, 1024, 32, w1aF);
  tw = 5  * NT * 512; pack_w_kernel<<<(tw + 255) / 256, 256, 0, stream>>>(a_w2, HID,  5,  w2aF);
  tw = 96 * NT * 512; pack_w_kernel<<<(tw + 255) / 256, 256, 0, stream>>>(s_w1, 3072, 96, w1sF);
  tw = 5  * NT * 512; pack_w_kernel<<<(tw + 255) / 256, 256, 0, stream>>>(s_w2, HID,  5,  w2sF);

  // ---- attention score MLP over 16384 tokens ----
  gemm_l1_kernel<<<1024, GEMM_BLOCK, 0, stream>>>(embeds, 1024, 32, w1aF, a_b1, h1F);
  gemm_l2_kernel<<<1024, GEMM_BLOCK, 0, stream>>>(h1F, w2aF, a_b2, h2);
  dot_l3_kernel<<<(16384 + 255) / 256, 256, 0, stream>>>(h2, a_w3, a_b3, scoreA, 16384);

  // ---- prefix sums of attention-weighted embeds ----
  prefix_partial_kernel<<<256, 256, 0, stream>>>(embeds, scoreA, csum);
  prefix_scan_kernel<<<32, 256, 0, stream>>>(csum);
  prefix_write_kernel<<<256, 256, 0, stream>>>(embeds, scoreA, csum, pre);

  // ---- span gather -> span_embeds (first 49,152,000 floats of out) ----
  gather_kernel<<<16000, 256, 0, stream>>>(embeds, pre, starts, lens, out);

  // ---- mention score MLP over 16000 spans (A staged from f32 out) ----
  gemm_l1_kernel<<<1000, GEMM_BLOCK, 0, stream>>>(out, 3072, 96, w1sF, s_b1, h1F);
  gemm_l2_kernel<<<1000, GEMM_BLOCK, 0, stream>>>(h1F, w2sF, s_b2, h2);
  dot_l3_kernel<<<(16000 + 255) / 256, 256, 0, stream>>>(h2, s_w3, s_b3,
                                                         out + 49152000ULL, 16000);
}
